// ChildSumTreeLSTMCell_61272003445201
// MI455X (gfx1250) — matrix operand order repaired and checked by reading, not compiled, not run on hardware
//
#include <hip/hip_runtime.h>
#include <hip/hip_bf16.h>

// ---------------------------------------------------------------------------
// Child-Sum TreeLSTM on the implicit 8-ary heap tree, CDNA5 (gfx1250).
//   N=65536, X=256, H=128. parent(i)=(i-1)/8. Node-level ranges:
//     L0=[8192,65536) L1=[1024,8192) L2=[128,1024) L3=[16,128) L4=[2,16)
//     L5=[1,2) L6=[0,1)
// GEMMs: V_WMMA_F32_16X16X32_F16 (f16 in, f32 accum).
// Recurrence A-tiles staged LDS-side with TENSOR_LOAD_TO_LDS (TDM, 6-arg
// clang-23 builtin) and consumed via ds_load; completion via
// s_wait_tensorcnt (per-wave TENSORcnt, no barrier needed).
// ---------------------------------------------------------------------------

typedef __attribute__((ext_vector_type(16))) _Float16 v16h;
typedef __attribute__((ext_vector_type(8)))  float    v8f;
typedef unsigned int u32x4 __attribute__((ext_vector_type(4)));
typedef int          i32x8 __attribute__((ext_vector_type(8)));
typedef int          i32x4 __attribute__((ext_vector_type(4)));

#define NNODES 65536
#define XS     256        // x feature size
#define HS     128        // hidden size
#define CW     512        // concatenated gate width: 384 (i|o|u) + 128 (f)

__device__ __forceinline__ float sigmoidf_(float v) {
    return 1.0f / (1.0f + __expf(-v));
}

// B fragment (32x16, f16): lane = column, 16 consecutive K per lane,
// lanes 16-31 hold K+16.  Source must be K-contiguous per column.
__device__ __forceinline__ v16h load_b_contig(const _Float16* __restrict__ bp) {
    v16h b;
#pragma unroll
    for (int i = 0; i < 16; ++i) b[i] = bp[i];
    return b;
}

// A fragment (16x32, f16) from f32 source row: lanes 0-15 hold
// K = {0..15}, lanes 16-31 hold K = {16..31} (kA = 0 / 16).
__device__ __forceinline__ v16h load_a_f32(const float* __restrict__ arow,
                                           int ko, int kA) {
    v16h a;
#pragma unroll
    for (int i = 0; i < 8; ++i) {
        a[i]     = (_Float16)arow[ko + kA + i];
        a[8 + i] = (_Float16)arow[ko + kA + 8 + i];
    }
    return a;
}

// ---------------------------------------------------------------------------
// K0: weight conversion to f16, both matrices K-contiguous per output column:
//   wcat [j*256+k] = (j<384 ? W_iou[j,k] : W_f[j-384,k])
//   ucatT[j*128+k] = (j<384 ? U_iou[k,j] : U_f[k,j-384])
// ---------------------------------------------------------------------------
__global__ __launch_bounds__(256) void k_convert_weights(
    const float* __restrict__ W_iou, const float* __restrict__ W_f,
    const float* __restrict__ U_iou, const float* __restrict__ U_f,
    _Float16* __restrict__ wcat, _Float16* __restrict__ ucatT)
{
    int tid = blockIdx.x * 256 + threadIdx.x;
    if (tid < CW * XS) {
        int j = tid >> 8, k = tid & 255;
        float v = (j < 384) ? W_iou[j * XS + k] : W_f[(j - 384) * XS + k];
        wcat[j * XS + k] = (_Float16)v;
    } else if (tid < CW * XS + CW * HS) {
        int t = tid - CW * XS;
        int j = t >> 7, k = t & 127;
        float v = (j < 384) ? U_iou[k * 384 + j] : U_f[k * HS + (j - 384)];
        ucatT[j * HS + k] = (_Float16)v;
    }
}

// ---------------------------------------------------------------------------
// K1: x_pre[n,0:512] = x @ Wcat^T + bias.  One wave -> 16 rows x 64 cols.
// Double-buffered K loop so loads for step k+1 overlap WMMAs of step k.
// ---------------------------------------------------------------------------
__global__ __launch_bounds__(256) void k_precompute_gemm(
    const float* __restrict__ x, const _Float16* __restrict__ wcat,
    const float* __restrict__ b_iou, const float* __restrict__ b_f,
    float* __restrict__ xpre)
{
    const int lane = threadIdx.x & 31;
    const int wave = threadIdx.x >> 5;
    const int gw   = blockIdx.x * 8 + wave;     // 0..32767
    const int rowtile = gw >> 3;                // 0..4095
    const int col0    = (gw & 7) * 64;
    const int m    = lane & 15;
    const int half = lane >> 4;
    const int kA   = half ? 16 : 0;
    const int kB   = half ? 16 : 0;
    const float* arow = x + (size_t)(rowtile * 16 + m) * XS;

    v8f acc[4];
#pragma unroll
    for (int t = 0; t < 4; ++t) acc[t] = (v8f){};

    v16h af[2], bf[2][4];
    af[0] = load_a_f32(arow, 0, kA);
#pragma unroll
    for (int t = 0; t < 4; ++t)
        bf[0][t] = load_b_contig(wcat + (size_t)(col0 + t * 16 + m) * XS + kB);

#pragma unroll
    for (int kk = 0; kk < 8; ++kk) {            // K = 256 = 8 x 32
        const int cur = kk & 1, nxt = cur ^ 1;
        if (kk < 7) {
            const int ko = (kk + 1) * 32;
            af[nxt] = load_a_f32(arow, ko, kA);
#pragma unroll
            for (int t = 0; t < 4; ++t)
                bf[nxt][t] = load_b_contig(
                    wcat + (size_t)(col0 + t * 16 + m) * XS + ko + kB);
        }
#pragma unroll
        for (int t = 0; t < 4; ++t)
            acc[t] = __builtin_amdgcn_wmma_f32_16x16x32_f16(
                false, af[cur], false, bf[cur][t], (short)0, acc[t], false, false);
    }
    // D layout: lanes 0-15 -> rows r, lanes 16-31 -> rows 8+r, col = lane%16
#pragma unroll
    for (int t = 0; t < 4; ++t) {
        const int col = col0 + t * 16 + m;
        const float bias = (col < 384) ? b_iou[col] : b_f[col - 384];
#pragma unroll
        for (int r = 0; r < 8; ++r) {
            const int row = rowtile * 16 + r + 8 * half;
            xpre[(size_t)row * CW + col] = acc[t][r] + bias;
        }
    }
}

// ---------------------------------------------------------------------------
// K2: leaf gates, nodes [8192, 65536): no children -> child sums are zero.
// ---------------------------------------------------------------------------
__global__ __launch_bounds__(256) void k_leaf_gates(
    const float* __restrict__ xpre, float* __restrict__ h_out,
    float* __restrict__ c_out, _Float16* __restrict__ h16)
{
    int idx = blockIdx.x * 256 + threadIdx.x;
    if (idx >= 57344 * HS) return;
    int node = 8192 + idx / HS;
    int j = idx & (HS - 1);
    const float* xp = xpre + (size_t)node * CW;
    float cn = sigmoidf_(xp[j]) * tanhf(xp[256 + j]);
    float hn = sigmoidf_(xp[128 + j]) * tanhf(cn);
    c_out[(size_t)node * HS + j] = cn;
    h_out[(size_t)node * HS + j] = hn;
    h16 [(size_t)node * HS + j] = (_Float16)hn;
}

// ---------------------------------------------------------------------------
// K3: one recurrence level; parents p in [n0,n1). Wave owns 2 parents ->
// child rows 8*p0+1 .. 8*p0+16 = one contiguous [16 x 128] f16 tile, staged
// into LDS by the Tensor Data Mover.  D# per ISA §8.3/8.4:
//   group0: count=1 | lds_addr | global_addr(57b) | type=2
//   group1: data_size=2B, tensor_dim0=128, tensor_dim1=N-(8p0+1) (clips the
//           missing child of node 8191 to zero via TDM OOB rules),
//           tile=128x16, tensor_dim0_stride=128
// Completion: s_wait_tensorcnt 0 (per-wave counter, no barrier needed).
// ---------------------------------------------------------------------------
__global__ __launch_bounds__(256) void k_level(
    const _Float16* __restrict__ ucatT, const _Float16* __restrict__ h16c,
    const float* __restrict__ xpre, float* __restrict__ h_out,
    float* __restrict__ c_out, _Float16* __restrict__ h16, int n0, int n1)
{
    __shared__ _Float16 s_a [8][16 * HS];  // 32 KB: per-wave TDM-staged A tile
    __shared__ float    s_uh[8][2][384];   // 24 KB: per-parent Uh sums
    __shared__ float    s_fc[8][2][128];   //  8 KB: per-parent fc sums

    const int lane = threadIdx.x & 31;
    const int wave = threadIdx.x >> 5;
    const int p0 = n0 + (blockIdx.x * 8 + wave) * 2;   // wave-uniform
    if (p0 >= n1) return;                              // uniform exit

    // ---- TDM: stage h16 rows [8p0+1, 8p0+17) into s_a[wave] ----
    {
        const unsigned s_first = 8u * (unsigned)p0 + 1u;
        const unsigned rows    = (unsigned)NNODES - s_first;   // clip OOB rows
        const unsigned long long ga =
            (unsigned long long)(const void*)(h16c + (size_t)s_first * HS);
        const unsigned ldsoff = (unsigned)(size_t)(void*)&s_a[wave][0];

        u32x4 g0;
        g0[0] = 1u;                                        // count=1, user mode
        g0[1] = ldsoff;                                    // lds_addr (bytes)
        g0[2] = (unsigned)(ga & 0xffffffffu);              // global_addr lo
        g0[3] = (unsigned)((ga >> 32) & 0x01ffffffu) | (2u << 30); // hi | type=2

        i32x8 g1 = (i32x8){};
        g1[0] = 0x00010000;                 // workgroup_mask=0, data_size=1 (2B)
        g1[1] = (int)(128u << 16);          // tensor_dim0[15:0]=128 (bits 63:48)
        g1[2] = (int)((rows & 0xffffu) << 16);        // tdim0 hi=0 | tdim1 lo
        g1[3] = (int)((rows >> 16) | (128u << 16));   // tdim1 hi | tile_dim0=128
        g1[4] = 16;                         // tile_dim1=16, tile_dim2=0
        g1[5] = 128;                        // tensor_dim0_stride = 128 elems
        g1[6] = 0; g1[7] = 0;               // (stride hi / dim1_stride unused)

        i32x4 g2 = (i32x4){};               // group2: unused (2-D tensor)
        i32x4 g3 = (i32x4){};               // group3: unused (2-D tensor)
        i32x8 g4 = (i32x8){};               // extra group of 6-arg builtin
        __builtin_amdgcn_tensor_load_to_lds(g0, g1, g2, g3, g4, 0);
        __builtin_amdgcn_s_wait_tensorcnt(0);
    }

    const int m    = lane & 15;
    const int half = lane >> 4;
    const int kA   = half ? 16 : 0;
    const int kB   = half ? 16 : 0;

    // A fragments for all 4 K-steps from the LDS tile (row m = child row)
    const _Float16* aw = &s_a[wave][m * HS];
    v16h af[4];
#pragma unroll
    for (int kk = 0; kk < 4; ++kk) {
        v16h a;
        const _Float16* p = aw + kk * 32 + kA;
#pragma unroll
        for (int i = 0; i < 8; ++i) { a[i] = p[i]; a[8 + i] = p[8 + i]; }
        af[kk] = a;
    }

    for (int jt = 0; jt < 32; ++jt) {                  // 32 column tiles of 16
        // B fragments: ucatT is K-contiguous per column -> b128 loads
        v16h bt[4];
#pragma unroll
        for (int kk = 0; kk < 4; ++kk)
            bt[kk] = load_b_contig(
                ucatT + (size_t)(jt * 16 + m) * HS + kk * 32 + kB);
        v8f acc = (v8f){};
#pragma unroll
        for (int kk = 0; kk < 4; ++kk)
            acc = __builtin_amdgcn_wmma_f32_16x16x32_f16(
                false, af[kk], false, bt[kk], (short)0, acc, false, false);

        if (jt < 24) {
            // Uh_sum: rows 0-7 (lanes 0-15, parent A) / rows 8-15 (parent B)
            float su = 0.f;
#pragma unroll
            for (int r = 0; r < 8; ++r) su += acc[r];
            s_uh[wave][half][jt * 16 + m] = su;
        } else {
            // fc_sum = sum_r c[child_r] * sigmoid(x_f + (h U_f))
            const int colf = (jt - 24) * 16 + m;
            const int pp = p0 + half;
            const bool pv = pp < n1;
            const float xf = pv ? xpre[(size_t)pp * CW + 384 + colf] : 0.f;
            float fcs = 0.f;
#pragma unroll
            for (int r = 0; r < 8; ++r) {
                const int sc = 8 * pp + 1 + r;
                const float cc = (pv && sc < NNODES)
                                     ? c_out[(size_t)sc * HS + colf] : 0.f;
                fcs += cc * sigmoidf_(xf + acc[r]);
            }
            s_fc[wave][half][colf] = fcs;
        }
    }

    // gates: 2 parents x 128 cols = 8 elements/lane (wave-private LDS)
#pragma unroll
    for (int t = 0; t < 8; ++t) {
        const int idx = t * 32 + lane;
        const int hh = idx >> 7, j = idx & 127;
        const int pp = p0 + hh;
        if (pp < n1) {
            const float* xp = xpre + (size_t)pp * CW;
            float iv = xp[j]       + s_uh[wave][hh][j];
            float ov = xp[128 + j] + s_uh[wave][hh][128 + j];
            float uv = xp[256 + j] + s_uh[wave][hh][256 + j];
            float cn = sigmoidf_(iv) * tanhf(uv) + s_fc[wave][hh][j];
            float hn = sigmoidf_(ov) * tanhf(cn);
            c_out[(size_t)pp * HS + j] = cn;
            h_out[(size_t)pp * HS + j] = hn;
            h16 [(size_t)pp * HS + j] = (_Float16)hn;
        }
    }
}

// ---------------------------------------------------------------------------
// Launcher. ws: x_pre [N,512] f32 (128MB) | h16 [N,128] f16 (16MB) |
//              wcat [512,256] f16 | ucatT [512,128] f16   (~144.4 MB total)
// d_out = h [N,128] f32 then c [N,128] f32 (reference returns (h, c)).
// ---------------------------------------------------------------------------
extern "C" void kernel_launch(void* const* d_in, const int* in_sizes, int n_in,
                              void* d_out, int out_size, void* d_ws, size_t ws_size,
                              hipStream_t stream) {
    (void)in_sizes; (void)n_in; (void)out_size; (void)ws_size;
    const float* x     = (const float*)d_in[0];
    const float* W_iou = (const float*)d_in[1];
    const float* U_iou = (const float*)d_in[2];
    const float* b_iou = (const float*)d_in[3];
    const float* W_f   = (const float*)d_in[4];
    const float* U_f   = (const float*)d_in[5];
    const float* b_f   = (const float*)d_in[6];
    // d_in[7..11] (edge arrays, node_level, num_levels) are implied by the
    // fixed 8-ary heap structure and not needed on device.

    float* h_out = (float*)d_out;
    float* c_out = h_out + (size_t)NNODES * HS;

    char* ws = (char*)d_ws;
    const size_t XPRE_BYTES = (size_t)NNODES * CW * sizeof(float);
    const size_t H16_BYTES  = (size_t)NNODES * HS * sizeof(_Float16);
    const size_t WCAT_BYTES = (size_t)CW * XS * sizeof(_Float16);
    float*    xpre  = (float*)ws;
    _Float16* h16   = (_Float16*)(ws + XPRE_BYTES);
    _Float16* wcat  = (_Float16*)(ws + XPRE_BYTES + H16_BYTES);
    _Float16* ucatT = (_Float16*)(ws + XPRE_BYTES + H16_BYTES + WCAT_BYTES);

    k_convert_weights<<<(CW * XS + CW * HS + 255) / 256, 256, 0, stream>>>(
        W_iou, W_f, U_iou, U_f, wcat, ucatT);

    k_precompute_gemm<<<4096, 256, 0, stream>>>(x, wcat, b_iou, b_f, xpre);

    k_leaf_gates<<<(57344 * HS + 255) / 256, 256, 0, stream>>>(
        xpre, h_out, c_out, h16);

    const int lv0[6] = {1024, 128, 16, 2, 1, 0};
    const int lv1[6] = {8192, 1024, 128, 16, 2, 1};
    for (int l = 0; l < 6; ++l) {
        const int P = lv1[l] - lv0[l];
        const int blocks = (P + 15) / 16;
        k_level<<<blocks, 256, 0, stream>>>(
            ucatT, h16, xpre, h_out, c_out, h16, lv0[l], lv1[l]);
    }
}
